// SupervisedContrastiveLoss_35948876268190
// MI455X (gfx1250) — matrix operand order repaired and checked by reading, not compiled, not run on hardware
//
#include <hip/hip_runtime.h>
#include <hip/hip_fp16.h>
#include <hip/hip_bf16.h>

typedef __attribute__((ext_vector_type(16))) _Float16 v16h;
typedef __attribute__((ext_vector_type(8)))  float    v8f;

#define HDIM 256
#define INV_T 14.2857142857142857f   // 1/0.07
#define MB    64                      // rows per block in main kernel
#define NSPLIT 8                      // column splits (grid.y)

static constexpr int LROW = HDIM + 8; // padded LDS row stride in halves (264 -> 528B, 16B aligned)

// ---------------- Kernel 1: L2-normalize rows, emit f16 ----------------
__global__ __launch_bounds__(256)
void normalize_kernel(const float* __restrict__ emb, _Float16* __restrict__ ehalf, int n) {
    int wave = (blockIdx.x * blockDim.x + threadIdx.x) >> 5;
    int lane = threadIdx.x & 31;
    if (wave >= n) return;
    const float4* row = (const float4*)(emb + (size_t)wave * HDIM);
    float4 x0 = row[lane * 2 + 0];
    float4 x1 = row[lane * 2 + 1];
    float ss = x0.x*x0.x + x0.y*x0.y + x0.z*x0.z + x0.w*x0.w
             + x1.x*x1.x + x1.y*x1.y + x1.z*x1.z + x1.w*x1.w;
    #pragma unroll
    for (int m = 16; m >= 1; m >>= 1) ss += __shfl_xor(ss, m, 32);
    float scale = 1.0f / fmaxf(sqrtf(ss), 1e-12f);
    union { _Float16 h[8]; uint4 u; } o;
    o.h[0] = (_Float16)(x0.x * scale); o.h[1] = (_Float16)(x0.y * scale);
    o.h[2] = (_Float16)(x0.z * scale); o.h[3] = (_Float16)(x0.w * scale);
    o.h[4] = (_Float16)(x1.x * scale); o.h[5] = (_Float16)(x1.y * scale);
    o.h[6] = (_Float16)(x1.z * scale); o.h[7] = (_Float16)(x1.w * scale);
    ((uint4*)(ehalf + (size_t)wave * HDIM))[lane] = o.u;
}

// ---------------- Kernel 1b: zero accumulators ----------------
__global__ void zero_kernel(float* p, int n) {
    int i = blockIdx.x * blockDim.x + threadIdx.x;
    if (i < n) p[i] = 0.0f;
}

// ---------------- B-tile stager: 16 rows x 512B into LDS ----------------
__device__ __forceinline__
void stage_b(_Float16* dstbase, const _Float16* __restrict__ ehalf, int nbase, int tid) {
    int row   = tid >> 3;   // 0..15
    int chunk = tid & 7;    // 0..7, 64B each
    const uint4* src = (const uint4*)(ehalf + (size_t)(nbase + row) * HDIM) + chunk * 4;
    uint4* dst = (uint4*)(dstbase + row * LROW + chunk * 32);
    #pragma unroll
    for (int i = 0; i < 4; ++i) dst[i] = src[i];
}

// ---------------- Kernel 2: fused GEMM (WMMA) + contrastive statistics ----------------
__global__ __launch_bounds__(128)
void scl_main(const _Float16* __restrict__ ehalf, const int* __restrict__ labels,
              float* __restrict__ sAcc, float* __restrict__ tAcc, float* __restrict__ cAcc,
              int n) {
    __shared__ __align__(16) _Float16 lds_a[MB][LROW];
    __shared__ __align__(16) _Float16 lds_b[2][16][LROW];
    __shared__ int lds_lab[4096];

    const int tid    = threadIdx.x;
    const int lane   = tid & 31;
    const int wave   = tid >> 5;
    const int mblock = blockIdx.x * MB;
    const int colsPer = n / NSPLIT;
    const int nbase0  = blockIdx.y * colsPer;
    const int nIter   = colsPer / 16;

    // Stage A block (64 rows, two 128B pieces per row) into LDS; stage labels
    {
        int row = tid >> 1, half = tid & 1;
        const uint4* src = (const uint4*)(ehalf + (size_t)(mblock + row) * HDIM) + half * 8;
        uint4* dst = (uint4*)(&lds_a[row][half * 128]);
        #pragma unroll
        for (int i = 0; i < 8; ++i) dst[i] = src[i];
    }
    for (int j = tid; j < n; j += 128) lds_lab[j] = labels[j];
    __syncthreads();

    const int hi    = lane >> 4;   // half-wave select (ISA fragment layout)
    const int l15   = lane & 15;
    const int mtile = wave * 16;   // this wave's M-tile inside lds_a

    int labI[8];
    #pragma unroll
    for (int v = 0; v < 8; ++v) labI[v] = lds_lab[mblock + mtile + v + hi * 8];

    // Hoist the wave's loop-invariant A fragments (16x256 tile) into registers:
    // removes 16 ds_load_b128 per iteration from the hot loop (64 VGPRs).
    v16h aFrag[8];
    #pragma unroll
    for (int ks = 0; ks < 8; ++ks) {
        const int kk = ks * 32;
        const uint4* pa = (const uint4*)(&lds_a[mtile + l15][kk + hi * 8]);
        ((uint4*)&aFrag[ks])[0] = pa[0];
        ((uint4*)&aFrag[ks])[1] = pa[2];   // +16 halves = +2 uint4
    }

    float sA[8], tA[8], cA[8];
    #pragma unroll
    for (int v = 0; v < 8; ++v) { sA[v] = 0.f; tA[v] = 0.f; cA[v] = 0.f; }

    for (int it = 0; it < nIter; ++it) {
        const int nbase = nbase0 + it * 16;
        if (it == 0) {
            stage_b(&lds_b[0][0][0], ehalf, nbase, tid);
            __syncthreads();
        }
        if (it + 1 < nIter) stage_b(&lds_b[(it + 1) & 1][0][0], ehalf, nbase + 16, tid);
        if (it + 2 < nIter) {
            const char* pf = (const char*)(ehalf + (size_t)(nbase + 32) * HDIM);
            __builtin_prefetch(pf + tid * 64, 0, 1);  // global_prefetch_b8, 8KB tile
        }

        // ---- 16x16x256 via 8 x v_wmma_f32_16x16x32_f16 (A in registers) ----
        const int buf = it & 1;
        v8f acc = {0.f, 0.f, 0.f, 0.f, 0.f, 0.f, 0.f, 0.f};
        #pragma unroll
        for (int ks = 0; ks < 8; ++ks) {
            const int kk = ks * 32;
            v16h b;
            // B frag: col row = l15, halves kk + hi*8 + [0..7] and kk + 16 + hi*8 + [0..7]
            const uint4* pb = (const uint4*)(&lds_b[buf][l15][kk + hi * 8]);
            ((uint4*)&b)[0] = pb[0];
            ((uint4*)&b)[1] = pb[2];
            acc = __builtin_amdgcn_wmma_f32_16x16x32_f16(
                false, aFrag[ks], false, b, (short)0, acc, false, false);
        }

        // ---- fused epilogue on C fragment: lane=col N=l15, VGPR v = row M=v+hi*8 ----
        const int j    = nbase + l15;
        const int labJ = lds_lab[j];
        #pragma unroll
        for (int v = 0; v < 8; ++v) {
            const int i = mblock + mtile + v + hi * 8;
            const float x = acc[v] * INV_T;            // sim value
            const bool diag = (i == j);
            const bool pos  = (!diag) && (labJ == labI[v]);
            sA[v] += diag ? 0.0f : __expf(x - INV_T);  // fixed shift m = 1/T (exact row max)
            if (pos) { tA[v] += x; cA[v] += 1.0f; }
        }
        __syncthreads();  // protect double-buffered B across iterations
    }

    // Reduce across the 16 columns held by each 16-lane group
    #pragma unroll
    for (int v = 0; v < 8; ++v) {
        #pragma unroll
        for (int m = 8; m >= 1; m >>= 1) {
            sA[v] += __shfl_xor(sA[v], m, 16);
            tA[v] += __shfl_xor(tA[v], m, 16);
            cA[v] += __shfl_xor(cA[v], m, 16);
        }
    }
    if (l15 == 0) {
        #pragma unroll
        for (int v = 0; v < 8; ++v) {
            const int i = mblock + mtile + v + hi * 8;
            atomicAdd(&sAcc[i], sA[v]);
            atomicAdd(&tAcc[i], tA[v]);
            atomicAdd(&cAcc[i], cA[v]);
        }
    }
}

// ---------------- Kernel 3: per-row loss + global reduction ----------------
__global__ __launch_bounds__(256)
void finalize_kernel(const float* __restrict__ sAcc, const float* __restrict__ tAcc,
                     const float* __restrict__ cAcc, float* __restrict__ out, int n) {
    __shared__ float redL[256];
    __shared__ float redV[256];
    float ls = 0.f, vs = 0.f;
    for (int i = threadIdx.x; i < n; i += 256) {
        float s = sAcc[i], t = tAcc[i], c = cAcc[i];
        // mean_log_prob_pos = [ (t - c*m) - c*log(s + 1e-8) ] / (c + 1e-8), m = 1/T
        float mean = ((t - c * INV_T) - c * logf(s + 1e-8f)) / (c + 1e-8f);
        if (c > 0.f) { ls += -mean; vs += 1.f; }
    }
    redL[threadIdx.x] = ls; redV[threadIdx.x] = vs;
    __syncthreads();
    for (int st = 128; st > 0; st >>= 1) {
        if (threadIdx.x < st) {
            redL[threadIdx.x] += redL[threadIdx.x + st];
            redV[threadIdx.x] += redV[threadIdx.x + st];
        }
        __syncthreads();
    }
    if (threadIdx.x == 0)
        out[0] = (redL[0] / fmaxf(redV[0], 1.0f)) * (0.07f / 0.07f);
}

// ---------------- Launch ----------------
extern "C" void kernel_launch(void* const* d_in, const int* in_sizes, int n_in,
                              void* d_out, int out_size, void* d_ws, size_t ws_size,
                              hipStream_t stream) {
    const float* emb    = (const float*)d_in[0];
    const int*   labels = (const int*)d_in[1];
    float*       out    = (float*)d_out;
    const int n = in_sizes[1];  // 4096 rows total

    _Float16* ehalf = (_Float16*)d_ws;
    float* sAcc = (float*)((char*)d_ws + (size_t)n * HDIM * sizeof(_Float16));
    float* tAcc = sAcc + n;
    float* cAcc = tAcc + n;

    normalize_kernel<<<dim3((n + 7) / 8), 256, 0, stream>>>(emb, ehalf, n);
    zero_kernel<<<dim3((3 * n + 255) / 256), 256, 0, stream>>>(sAcc, 3 * n);

    dim3 grid(n / MB, NSPLIT);
    scl_main<<<grid, 128, 0, stream>>>(ehalf, labels, sAcc, tAcc, cAcc, n);

    finalize_kernel<<<1, 256, 0, stream>>>(sAcc, tAcc, cAcc, out, n);
}
